// LinearChainCRF_15427522527483
// MI455X (gfx1250) — matrix operand order repaired and checked by reading, not compiled, not run on hardware
//
#include <hip/hip_runtime.h>
#include <hip/hip_bf16.h>

// Linear-chain CRF log-likelihood for MI455X (gfx1250, wave32).
//
//   B=256 chains, T=2048 steps, K=48 states.
//
//   logZ: the forward scan is chunked via the associative log-matmul semiring
//   (NC=8 chunks -> 48x48 transfer matrices, 256*8*3 = 6144 waves), and each
//   chunk step runs in LINEAR space with a per-column log-scale accumulator:
//       P' = exp(em_t) o (E^T x P),   E = exp(trans)   (bf16 WMMA)
//   renormalizing P (one v_log + v_rcp + 24 muls) only every 4 steps.
//   exp(emissions) is precomputed once (streaming kernel) so the serial inner
//   loop has ZERO transcendentals.
//
// Workspace: [0,1KB)            gold scores (256 f32)
//            [+18.9MB)          chunk matrices S[b][c][src][dst] (log space)
//            [+100.7MB)         exp(emissions) f32

#define NB 256
#define NT 2048
#define NK 48
#define NC 8            // time chunks
#define CH (NT / NC)    // 256 steps per chunk

#define WS_S_OFF  1024
#define WS_EE_OFF (1024 + (size_t)NB * NC * NK * NK * 4)

typedef __attribute__((ext_vector_type(16))) __bf16 v16bf;
typedef __attribute__((ext_vector_type(2)))  __bf16 v2bf;
typedef __attribute__((ext_vector_type(2)))  float  v2f;
typedef __attribute__((ext_vector_type(8)))  float  v8f;
typedef __attribute__((ext_vector_type(8)))  unsigned int v8u;

__device__ __forceinline__ float fast_exp(float x) {
  return __builtin_amdgcn_exp2f(x * 1.4426950408889634f);
}
__device__ __forceinline__ float fast_log(float x) {
  return __builtin_amdgcn_logf(x) * 0.6931471805599453f;
}

__device__ __forceinline__ v8f wmma_bf16(v16bf a, v16bf b, v8f c) {
  // (neg_a, A, neg_b, B, c_mod, C, reuse_a, reuse_b)
  return __builtin_amdgcn_wmma_f32_16x16x32_bf16(false, a, false, b,
                                                 (short)0, c, false, false);
}

// ---------------------------------------------------------------------------
// Stage -1: exp(emissions), streaming, float4-vectorized. 25.2M elements.
// ---------------------------------------------------------------------------
__global__ __launch_bounds__(256) void crf_expem_kernel(
    const float* __restrict__ em, float* __restrict__ ee)
{
  const size_t i = ((size_t)blockIdx.x * 256 + threadIdx.x) * 4;
  const float4 x = *(const float4*)(em + i);
  float4 y;
  y.x = fast_exp(x.x); y.y = fast_exp(x.y);
  y.z = fast_exp(x.z); y.w = fast_exp(x.w);
  *(float4*)(ee + i) = y;
}

// ---------------------------------------------------------------------------
// Stage 0: gold path score (gather + block reduction), one block per batch.
// ---------------------------------------------------------------------------
__global__ __launch_bounds__(256) void crf_gold_kernel(
    const float* __restrict__ em, const long long* __restrict__ tags,
    const unsigned char* __restrict__ mask,
    const float* __restrict__ startT, const float* __restrict__ endT,
    const float* __restrict__ trans, float* __restrict__ gold)
{
  const int b = blockIdx.x;
  const int tid = threadIdx.x;
  float s = 0.0f;
  int cnt = 0;
  for (int t = tid; t < NT; t += 256) {
    const int tg = (int)tags[(size_t)b * NT + t];
    const unsigned char mk = mask[(size_t)b * NT + t];
    cnt += mk ? 1 : 0;
    if (t >= 1 && mk) {
      const int tp = (int)tags[(size_t)b * NT + t - 1];
      s += em[((size_t)b * NT + t) * NK + tg] + trans[tp * NK + tg];
    }
  }
  __shared__ float sred[256];
  __shared__ int   cred[256];
  sred[tid] = s; cred[tid] = cnt;
  __syncthreads();
  for (int off = 128; off > 0; off >>= 1) {
    if (tid < off) { sred[tid] += sred[tid + off]; cred[tid] += cred[tid + off]; }
    __syncthreads();
  }
  if (tid == 0) {
    const int t0tag = (int)tags[(size_t)b * NT];
    float sc = sred[0] + startT[t0tag] + em[(size_t)b * NT * NK + t0tag];
    const int len = cred[0] - 1;                 // lengths = sum(mask) - 1
    const int lastTag = (int)tags[(size_t)b * NT + len];
    gold[b] = sc + endT[lastTag];
  }
}

// ---------------------------------------------------------------------------
// Stage 1: per (batch, time-chunk, 16-source-state group) wave computes the
// 48x48 log-space transfer matrix of the chunk.  State = (M[s], P[j][s]) with
// score = M + log P;  P lives in the WMMA C/D register layout and refeeds the
// next step's B operand with only 12 half-wave dword shuffles (DS pipe).
// ---------------------------------------------------------------------------
__global__ __launch_bounds__(32) void crf_chunk_kernel(
    const float* __restrict__ ee,      // exp(emissions)
    const unsigned char* __restrict__ mask,
    const float* __restrict__ trans, float* __restrict__ Sout)
{
  const int lane = threadIdx.x;
  const int col  = lane & 15;          // column index within 16-wide group
  const bool hi  = lane >= 16;
  const int joff = hi ? 8 : 0;         // D-layout row offset for upper lanes
  const unsigned keep = hi ? 0u : 0xffffffffu;   // B1 zero padding via AND

  const int id = blockIdx.x;
  const int sg = id % 3;               // source-state column group (0..2)
  const int c  = (id / 3) % NC;        // time chunk
  const int b  = id / (3 * NC);        // batch
  const int s  = sg * 16 + col;        // this lane-column's source state

  // --- A operands (chunk-invariant): A[tile][half],
  // A[m][k] = E^T[j][i] = exp(trans[i][j]), j = tile*16 + (lane&15),
  // i = half*32 + k, zero padded for i >= 48.
  // 16-bit A 16x32, k order used here (contiguous): lanes 0-15: V0-7 K=0..15;
  //                        lanes 16-31: V0-7 K=16..31.
  v16bf A[3][2];
#pragma unroll
  for (int tile = 0; tile < 3; ++tile) {
    const int j = tile * 16 + col;
#pragma unroll
    for (int half = 0; half < 2; ++half) {
      v16bf a;
#pragma unroll
      for (int e = 0; e < 16; ++e) {
        const int v  = e >> 1;
        const int kk = (hi ? 16 : 0) + e + 0 * v;
        const int i  = half * 32 + kk;
        const float val = (i < NK) ? fast_exp(trans[i * NK + j]) : 0.0f;
        a[e] = (__bf16)val;
      }
      A[tile][half] = a;
    }
  }

  // --- linear-space identity init: P = I, M = 0
  float P[3][8];
#pragma unroll
  for (int tt = 0; tt < 3; ++tt)
#pragma unroll
    for (int r = 0; r < 8; ++r)
      P[tt][r] = (tt * 16 + joff + r == s) ? 1.0f : 0.0f;
  float M = 0.0f;

  const int tfirst = c * CH + 1;
  const int tlast  = ((c + 1) * CH < NT - 1) ? (c + 1) * CH : NT - 1;

  // Strength-reduced addressing: running pointers, no per-iteration mul.
  const float* eptr = ee + ((size_t)b * NT + tfirst) * NK;
  const unsigned char* mptr = mask + (size_t)b * NT + tfirst;

  for (int t = tfirst; t <= tlast; ++t, eptr += NK, ++mptr) {
    // Amortized renormalization: keep max(P) ~ 1 in the log accumulator.
    // Worst-case per-step growth ~1e4 -> 1e16 over 4 steps, safe in f32.
    if ((t & 3) == 0) {
      float cm = P[0][0];
#pragma unroll
      for (int tt = 0; tt < 3; ++tt)
#pragma unroll
        for (int r = 0; r < 8; ++r) cm = fmaxf(cm, P[tt][r]);
      cm = fmaxf(cm, __shfl_xor(cm, 16, 32));
      M += fast_log(cm);
      const float rs = __builtin_amdgcn_rcpf(cm);
#pragma unroll
      for (int tt = 0; tt < 3; ++tt)
#pragma unroll
        for (int r = 0; r < 8; ++r) P[tt][r] *= rs;
    }

    // exp(emissions) for this step in D layout (shared across the 16 cols).
    __builtin_prefetch(eptr + NK, 0, 1);   // next step -> global_prefetch_b8
    float eb[3][8];
#pragma unroll
    for (int tt = 0; tt < 3; ++tt) {
      const float4 x = *(const float4*)(eptr + tt * 16 + joff);
      const float4 y = *(const float4*)(eptr + tt * 16 + joff + 4);
      eb[tt][0] = x.x; eb[tt][1] = x.y; eb[tt][2] = x.z; eb[tt][3] = x.w;
      eb[tt][4] = y.x; eb[tt][5] = y.y; eb[tt][6] = y.z; eb[tt][7] = y.w;
    }
    const float mf = (float)(*mptr);       // uniform -> s_load_u8, branchless

    // Pack bf16 PAIRS (v_cvt_pk_bf16_f32 via vector fptrunc), then
    // shuffle/select dwords (12 DS-pipe shuffles).
    unsigned int pk[3][4], qk[3][4];
#pragma unroll
    for (int tt = 0; tt < 3; ++tt)
#pragma unroll
      for (int v = 0; v < 4; ++v) {
        const v2f pf = {P[tt][2 * v], P[tt][2 * v + 1]};
        pk[tt][v] = __builtin_bit_cast(unsigned int,
                                       __builtin_convertvector(pf, v2bf));
        qk[tt][v] = __shfl_xor(pk[tt][v], 16, 32);   // partner half's pair
      }

    // 16-bit B 32x16 in this code's contiguous k order (same as A):
    //  V0-7: lanes 0-15 = (col n, K=0..15), lanes 16-31 = (col n-16, K=16..31)
    //  half0 covers i=0..31, half1 covers i=32..63 (i>=48 zero padded).
    v8u u0, u1;
#pragma unroll
    for (int v = 0; v < 4; ++v) {
      u0[v]     = hi ? qk[1][v] : pk[0][v];  // lo: j=2v..   | hi: j=16+2v..
      u0[v + 4] = hi ? pk[1][v] : qk[0][v];  // lo: j=8+2v.. | hi: j=24+2v..
      u1[v]     = pk[2][v] & keep;           // lo: i=32+2v..| hi: pad
      u1[v + 4] = qk[2][v] & keep;           // lo: i=40+2v..| hi: pad
    }
    const v16bf B0 = __builtin_bit_cast(v16bf, u0);
    const v16bf B1 = __builtin_bit_cast(v16bf, u1);

    // All three B0-WMMAs, then all three B1-WMMAs: identical B operand on
    // consecutive instructions and a 3-instruction gap in each C->D chain.
    v8f acc[3];
#pragma unroll
    for (int tt = 0; tt < 3; ++tt) {
      v8f z = {};
      acc[tt] = wmma_bf16(A[tt][0], B0, z);
    }
#pragma unroll
    for (int tt = 0; tt < 3; ++tt)
      acc[tt] = wmma_bf16(A[tt][1], B1, acc[tt]);

    // P' = exp(em) o (E^T P);  P = P + mf*(P' - P)   (no branches, no logs)
#pragma unroll
    for (int tt = 0; tt < 3; ++tt)
#pragma unroll
      for (int r = 0; r < 8; ++r) {
        const float np = eb[tt][r] * acc[tt][r];
        P[tt][r] = __builtin_fmaf(mf, np - P[tt][r], P[tt][r]);
      }
  }

  // Back to log space once per chunk: S[b][c][source=s][dest=j] = M + log P.
  float* o = Sout + (((size_t)b * NC + c) * NK + s) * NK;
#pragma unroll
  for (int tt = 0; tt < 3; ++tt)
#pragma unroll
    for (int r = 0; r < 8; ++r)
      o[tt * 16 + joff + r] = M + fast_log(fmaxf(P[tt][r], 1e-37f));
}

// ---------------------------------------------------------------------------
// Stage 2: fold the NC chunk matrices into the start vector (log-matvec),
// then logZ = lse(v + end) and out = gold - logZ. One block per batch.
// ---------------------------------------------------------------------------
__global__ __launch_bounds__(64) void crf_combine_kernel(
    const float* __restrict__ em, const float* __restrict__ startT,
    const float* __restrict__ endT, const float* __restrict__ S,
    const float* __restrict__ gold, float* __restrict__ out)
{
  const int b = blockIdx.x;
  const int j = threadIdx.x;            // active for j < 48
  __shared__ float v[NK];
  __shared__ float nv[NK];
  if (j < NK) v[j] = startT[j] + em[(size_t)b * NT * NK + j];
  __syncthreads();
  for (int c = 0; c < NC; ++c) {
    const float* Sc = S + ((size_t)b * NC + c) * NK * NK;
    if (j < NK) {
      float mx = -1e30f;
      for (int i = 0; i < NK; ++i) mx = fmaxf(mx, v[i] + Sc[i * NK + j]);
      float sum = 0.0f;
      for (int i = 0; i < NK; ++i) sum += fast_exp(v[i] + Sc[i * NK + j] - mx);
      nv[j] = mx + fast_log(sum);
    }
    __syncthreads();
    if (j < NK) v[j] = nv[j];
    __syncthreads();
  }
  if (j == 0) {
    float mx = -1e30f;
    for (int i = 0; i < NK; ++i) mx = fmaxf(mx, v[i] + endT[i]);
    float sum = 0.0f;
    for (int i = 0; i < NK; ++i) sum += fast_exp(v[i] + endT[i] - mx);
    out[b] = gold[b] - (mx + fast_log(sum));
  }
}

extern "C" void kernel_launch(void* const* d_in, const int* in_sizes, int n_in,
                              void* d_out, int out_size, void* d_ws, size_t ws_size,
                              hipStream_t stream) {
  const float*         em     = (const float*)d_in[0];          // (B,T,K) f32
  const long long*     tags   = (const long long*)d_in[1];      // (B,T) i64
  const unsigned char* mask   = (const unsigned char*)d_in[2];  // (B,T) bool
  const float*         startT = (const float*)d_in[3];          // (K,)
  const float*         endT   = (const float*)d_in[4];          // (K,)
  const float*         trans  = (const float*)d_in[5];          // (K,K)

  float* gold = (float*)d_ws;                            // 256 f32
  float* S    = (float*)((char*)d_ws + WS_S_OFF);        // B*NC*K*K f32
  float* EE   = (float*)((char*)d_ws + WS_EE_OFF);       // B*T*K f32

  crf_expem_kernel<<<(NB * NT * NK) / (256 * 4), 256, 0, stream>>>(em, EE);
  crf_gold_kernel<<<NB, 256, 0, stream>>>(em, tags, mask, startT, endT, trans, gold);
  crf_chunk_kernel<<<NB * NC * 3, 32, 0, stream>>>(EE, mask, trans, S);
  crf_combine_kernel<<<NB, 64, 0, stream>>>(em, startT, endT, S, gold, (float*)d_out);
}
